// Head_41077067219182
// MI455X (gfx1250) — matrix operand order repaired and checked by reading, not compiled, not run on hardware
//
#include <hip/hip_runtime.h>

// ---------------------------------------------------------------------------
// Single-head causal attention for MI455X (gfx1250, wave32, WMMA).
//   x:[B,T,D] f32, Wk/Wq/Wv:[D,H] f32  ->  out:[B,T,H] f32
//   B=4 T=4096 D=1024 H=64
// All GEMMs via v_wmma_f32_16x16x32_bf16 (f32 accumulation).
//  K0: weight transpose/convert.
//  K1: QKV projection, W^T staged through LDS with async-to-LDS double buffer.
//  K2: split-K flash attention (4 waves/tile), prefetch, LDS P-relayout.
// ---------------------------------------------------------------------------

#define BB 4
#define TT 4096
#define DD 1024
#define HH 64
#define NW 4             // waves per query tile in flash_attn
#define WSTRIDE 40       // padded LDS row stride (elems): bank-conflict-free

typedef __attribute__((ext_vector_type(16))) __bf16 v16bf;
typedef __attribute__((ext_vector_type(8)))  __bf16 v8bf;
typedef __attribute__((ext_vector_type(8)))  float  v8f;
typedef __attribute__((ext_vector_type(8)))  float  f32x8;
typedef __attribute__((ext_vector_type(4)))  int    v4i;

#if __has_builtin(__builtin_amdgcn_global_load_async_to_lds_b128)
#define HAVE_ASYNC_LDS 1
#define ASYNC_WAIT() asm volatile("s_wait_asynccnt 0x0" ::: "memory")
// builtin signature: (int4 addrspace(1)*, int4 addrspace(3)*, imm offset, cpol)
#define ASYNC_CP16(gsrc, ldst)                                              \
  __builtin_amdgcn_global_load_async_to_lds_b128(                           \
      (__attribute__((address_space(1))) v4i*)(void*)(gsrc),                \
      (__attribute__((address_space(3))) v4i*)(void*)(ldst), 0, 0)
#else
#define HAVE_ASYNC_LDS 0
#define ASYNC_WAIT()
#define ASYNC_CP16(gsrc, ldst) (*(v8bf*)(ldst) = *(const v8bf*)(gsrc))
#endif

__device__ __forceinline__ v8f zero8() {
  v8f z;
#pragma unroll
  for (int i = 0; i < 8; ++i) z[i] = 0.0f;
  return z;
}

__device__ __forceinline__ v16bf cat8(v8bf lo, v8bf hi) {
  return __builtin_shufflevector(lo, hi, 0,1,2,3,4,5,6,7,8,9,10,11,12,13,14,15);
}

__device__ __forceinline__ v8f wmma_bf16(v16bf a, v16bf b, v8f c) {
  return __builtin_amdgcn_wmma_f32_16x16x32_bf16(false, a, false, b,
                                                 (short)0, c, false, false);
}

// ---------------------------------------------------------------------------
// Kernel 0: W [D,H] f32  ->  Wt [3][H,D] bf16   (order: q, k, v)
// ---------------------------------------------------------------------------
__global__ void prep_weights(const float* __restrict__ Wk,
                             const float* __restrict__ Wq,
                             const float* __restrict__ Wv,
                             __bf16* __restrict__ wt) {
  int i = blockIdx.x * blockDim.x + threadIdx.x;   // 0 .. 3*H*D-1
  int m = i >> 16;                                  // matrix id (H*D = 65536)
  int r = i & 65535;
  int h = r >> 10;                                  // 0..63
  int d = r & 1023;                                 // 0..1023
  const float* src = (m == 0) ? Wq : (m == 1) ? Wk : Wv;
  wt[i] = (__bf16)src[d * HH + h];                  // wt[m][h][d]
}

// ---------------------------------------------------------------------------
// Kernel 1: QKV projection. 8 waves/WG, one 16-row x-tile per wave.
// W^T d-chunk (3 mats x 64 h x 32 d = 12KB) staged cooperatively into LDS,
// double-buffered with async-to-LDS so staging overlaps the 12 WMMAs.
// ---------------------------------------------------------------------------
__global__ __launch_bounds__(256) void qkv_proj(const float*  __restrict__ x,
                                                const __bf16* __restrict__ wt,
                                                __bf16* __restrict__ qb,
                                                __bf16* __restrict__ kb,
                                                __bf16* __restrict__ vt) {
  __shared__ __bf16 ldsW[2][192 * WSTRIDE] __attribute__((aligned(16)));

  const int tid  = threadIdx.x;
  const int lane = tid & 31;
  const int wave = tid >> 5;
  const int tile = blockIdx.x * 8 + wave;   // 0..1023
  const int rowbase = tile * 16;            // global row of x
  const int c    = lane & 15;
  const int half = lane >> 4;

  // Cooperative stage of one 32-wide d-chunk of all three W^T matrices.
  // 192 rows (m*64+h) x 32 elems; 768 x 16B transfers, 3 per thread.
  auto stage = [&](int buf, int d0) {
#pragma unroll
    for (int e = 0; e < 3; ++e) {
      const int gi   = e * 256 + tid;       // 0..767
      const int row  = gi >> 2;             // 0..191
      const int part = gi & 3;              // 16B quarter of the 64B row chunk
      const __bf16* src = wt + (size_t)row * DD + d0 + part * 8;
      __bf16* dst = &ldsW[buf][row * WSTRIDE + part * 8];
      ASYNC_CP16(src, dst);
    }
  };

  v8f aq[4], ak[4], av[4];
#pragma unroll
  for (int j = 0; j < 4; ++j) { aq[j] = zero8(); ak[j] = zero8(); av[j] = zero8(); }

  const float* xrow = x + (size_t)(rowbase + c) * DD;

  stage(0, 0);
  ASYNC_WAIT();
  __syncthreads();

  for (int it = 0; it < DD / 32; ++it) {
    const int d0  = it * 32;
    const int buf = it & 1;

    // Kick off staging of the next chunk; completes during this chunk's math.
    if (it + 1 < DD / 32) stage(buf ^ 1, d0 + 32);

    // A fragment: lane holds row m=c, K chunk {16*half..16*half+15}
    f32x8 f0 = *(const f32x8*)(xrow + d0 + 16 * half);
    f32x8 f1 = *(const f32x8*)(xrow + d0 + 16 * half + 8);
    if (d0 + 32 < DD)
      __builtin_prefetch(xrow + d0 + 32 + 8 * half, 0, 3);

    v16bf a;
#pragma unroll
    for (int i = 0; i < 8; ++i) { a[i] = (__bf16)f0[i]; a[i + 8] = (__bf16)f1[i]; }

    // B fragments from LDS (rows padded to WSTRIDE: conflict-free b128 reads)
#pragma unroll
    for (int j = 0; j < 4; ++j) {
      const int rq = 0 * 64 + 16 * j + c;
      const int rk = 1 * 64 + 16 * j + c;
      const int rv = 2 * 64 + 16 * j + c;
      const __bf16* pq = &ldsW[buf][rq * WSTRIDE + 16 * half];
      const __bf16* pk = &ldsW[buf][rk * WSTRIDE + 16 * half];
      const __bf16* pv = &ldsW[buf][rv * WSTRIDE + 16 * half];
      v16bf bq = cat8(*(const v8bf*)pq, *(const v8bf*)(pq + 8));
      v16bf bk = cat8(*(const v8bf*)pk, *(const v8bf*)(pk + 8));
      v16bf bv = cat8(*(const v8bf*)pv, *(const v8bf*)(pv + 8));
      aq[j] = wmma_bf16(a, bq, aq[j]);
      ak[j] = wmma_bf16(a, bk, ak[j]);
      av[j] = wmma_bf16(a, bv, av[j]);
    }

    if (it + 1 < DD / 32) {
      asm volatile("s_wait_dscnt 0x0" ::: "memory");  // frag reads retired
      ASYNC_WAIT();                                   // next buffer populated
      __syncthreads();
    }
  }

  // Epilogue: C layout (vgpr r -> row r+8*half, lane -> col c)
  const int b  = rowbase >> 12;     // /4096
  const int t0 = rowbase & 4095;
#pragma unroll
  for (int j = 0; j < 4; ++j) {
#pragma unroll
    for (int r = 0; r < 8; ++r) {
      const int m  = r + 8 * half;
      const int hh = 16 * j + c;
      qb[(size_t)(rowbase + m) * HH + hh] = (__bf16)aq[j][r];
      kb[(size_t)(rowbase + m) * HH + hh] = (__bf16)ak[j][r];
      vt[((size_t)(b * HH + hh)) * TT + (t0 + m)] = (__bf16)av[j][r];
    }
  }
}

// ---------------------------------------------------------------------------
// Kernel 2: split-K flash attention.
// Workgroup = NW waves per 16-query tile (grid = B*T/16). Wave w processes
// key blocks w, w+NW, ... Partial (m, l, O~) combined via LDS at the end.
// Per block: K loads -> V loads -> 4 S-WMMA -> online softmax -> P via LDS
// -> 4 PV-WMMA; next block's K/V prefetched (global_prefetch_b8).
// ---------------------------------------------------------------------------
__global__ __launch_bounds__(32 * NW) void flash_attn(
    const __bf16* __restrict__ qb, const __bf16* __restrict__ kb,
    const __bf16* __restrict__ vt, float* __restrict__ out) {
  __shared__ __bf16 pbuf[NW][16 * 32] __attribute__((aligned(64)));
  __shared__ float  oP[NW - 1][16][HH];   // partial O~ of waves 1..NW-1
  __shared__ float  mP[NW - 1][16];
  __shared__ float  lP[NW - 1][16];

  const int lane = threadIdx.x & 31;
  const int wv   = threadIdx.x >> 5;  // wave in workgroup: 0..NW-1
  const int c    = lane & 15;
  const int half = lane >> 4;
  const int qt   = blockIdx.x;        // 0..1023
  const int b    = qt >> 8;           // /256 (tiles per batch)
  const int g0   = qt << 4;           // global query row base
  const int tloc = g0 & 4095;         // query row base within batch

  // Q as two A fragments (h-chunks 0..31 and 32..63), kept in registers.
  const __bf16* qrow = qb + (size_t)(g0 + c) * HH;
  v16bf qa0 = cat8(*(const v8bf*)(qrow +      16 * half),
                   *(const v8bf*)(qrow + 8 + 16 * half));
  v16bf qa1 = cat8(*(const v8bf*)(qrow + 32 + 16 * half),
                   *(const v8bf*)(qrow + 40 + 16 * half));

  v8f o[4];
#pragma unroll
  for (int j = 0; j < 4; ++j) o[j] = zero8();
  float mrow[8], lrow[8];
#pragma unroll
  for (int r = 0; r < 8; ++r) { mrow[r] = -1e30f; lrow[r] = 0.0f; }

  const float cexp = 0.125f * 1.44269504f;   // scale * log2(e)
  const __bf16* kbase = kb + (size_t)b * TT * HH;
  const __bf16* vbase = vt + (size_t)b * HH * TT;
  const int nkb = (tloc + 16 + 31) >> 5;     // causal key-block count

  for (int kbk = wv; kbk < nkb; kbk += NW) {
    const int s0 = kbk * 32;

    // Warm next block's K/V cachelines while this block computes.
    const int nb = kbk + NW;
    if (nb < nkb) {
      const int sn = nb * 32;
      __builtin_prefetch(kbase + (size_t)(sn + lane) * HH, 0, 3);
      __builtin_prefetch(vbase + (size_t)lane        * TT + sn, 0, 3);
      __builtin_prefetch(vbase + (size_t)(32 + lane) * TT + sn, 0, 3);
    }

    // K fragments first (S-WMMA waits only on these) ...
    const __bf16* kr0 = kbase + (size_t)(s0 + c)      * HH + 16 * half;
    const __bf16* kr1 = kbase + (size_t)(s0 + 16 + c) * HH + 16 * half;
    v16bf bK00 = *(const v16bf*)(kr0);        // keys s0..s0+15,  h 0..31
    v16bf bK01 = *(const v16bf*)(kr0 + 32);   //                  h 32..63
    v16bf bK10 = *(const v16bf*)(kr1);        // keys s0+16..+31, h 0..31
    v16bf bK11 = *(const v16bf*)(kr1 + 32);

    // ... then V fragments: stay in flight through softmax.
    v16bf bV[4];
#pragma unroll
    for (int j = 0; j < 4; ++j)
      bV[j] = *(const v16bf*)(vbase + (size_t)(16 * j + c) * TT + s0 + 16 * half);

    v8f S0 = zero8(), S1 = zero8();
    S0 = wmma_bf16(qa0, bK00, S0);  S0 = wmma_bf16(qa1, bK01, S0);
    S1 = wmma_bf16(qa0, bK10, S1);  S1 = wmma_bf16(qa1, bK11, S1);

    // ---- causal mask + online softmax (per C-layout row r+8*half) ----
#pragma unroll
    for (int r = 0; r < 8; ++r) {
      const int trow = tloc + r + 8 * half;
      float v0 = (s0 + c      <= trow) ? S0[r] : -1e30f;
      float v1 = (s0 + 16 + c <= trow) ? S1[r] : -1e30f;

      float bm = fmaxf(v0, v1);
#pragma unroll
      for (int off = 1; off < 16; off <<= 1)
        bm = fmaxf(bm, __shfl_xor(bm, off, 32));

      const float mnew  = fmaxf(mrow[r], bm);
      const float alpha = exp2f((mrow[r] - mnew) * cexp);
      const float e0 = exp2f((v0 - mnew) * cexp);
      const float e1 = exp2f((v1 - mnew) * cexp);

      float rs = e0 + e1;
#pragma unroll
      for (int off = 1; off < 16; off <<= 1)
        rs += __shfl_xor(rs, off, 32);

      lrow[r] = lrow[r] * alpha + rs;
      mrow[r] = mnew;
#pragma unroll
      for (int j = 0; j < 4; ++j) o[j][r] *= alpha;

      // stage P (C layout) into this wave's LDS tile, row-major [16][32]
      const int m = r + 8 * half;
      pbuf[wv][m * 32 + c]      = (__bf16)e0;
      pbuf[wv][m * 32 + 16 + c] = (__bf16)e1;
    }

    asm volatile("s_wait_dscnt 0x0" ::: "memory");   // same-wave LDS RAW

    // Reload P as A fragment (row m=c; K chunk 16*half..16*half+15)
    v16bf pa = cat8(*(const v8bf*)(&pbuf[wv][c * 32 + 16 * half]),
                    *(const v8bf*)(&pbuf[wv][c * 32 + 16 * half + 8]));

#pragma unroll
    for (int j = 0; j < 4; ++j)
      o[j] = wmma_bf16(pa, bV[j], o[j]);
  }

  // ---- split-K combine: waves 1..NW-1 publish partials, wave 0 reduces ----
  if (wv != 0) {
#pragma unroll
    for (int r = 0; r < 8; ++r) {
      const int m = r + 8 * half;
#pragma unroll
      for (int j = 0; j < 4; ++j) oP[wv - 1][m][16 * j + c] = o[j][r];
      if (c == 0) { mP[wv - 1][m] = mrow[r]; lP[wv - 1][m] = lrow[r]; }
    }
  }
  __syncthreads();

  if (wv == 0) {
#pragma unroll
    for (int r = 0; r < 8; ++r) {
      const int m = r + 8 * half;
      float mstar = mrow[r];
#pragma unroll
      for (int w = 0; w < NW - 1; ++w) mstar = fmaxf(mstar, mP[w][m]);

      const float a0 = exp2f((mrow[r] - mstar) * cexp);
      float lsum = lrow[r] * a0;
      float acc[4];
#pragma unroll
      for (int j = 0; j < 4; ++j) acc[j] = o[j][r] * a0;

#pragma unroll
      for (int w = 0; w < NW - 1; ++w) {
        const float aw = exp2f((mP[w][m] - mstar) * cexp);
        lsum += lP[w][m] * aw;
#pragma unroll
        for (int j = 0; j < 4; ++j) acc[j] += oP[w][m][16 * j + c] * aw;
      }

      const float inv = 1.0f / lsum;
#pragma unroll
      for (int j = 0; j < 4; ++j)
        out[(size_t)(g0 + m) * HH + 16 * j + c] = acc[j] * inv;
    }
  }
}

// ---------------------------------------------------------------------------
// Launch
// ---------------------------------------------------------------------------
extern "C" void kernel_launch(void* const* d_in, const int* in_sizes, int n_in,
                              void* d_out, int out_size, void* d_ws, size_t ws_size,
                              hipStream_t stream) {
  (void)in_sizes; (void)n_in; (void)out_size; (void)ws_size;

  const float* x  = (const float*)d_in[0];
  const float* Wk = (const float*)d_in[1];
  const float* Wq = (const float*)d_in[2];
  const float* Wv = (const float*)d_in[3];
  float* out = (float*)d_out;

  char* ws = (char*)d_ws;
  __bf16* wt  = (__bf16*)ws;                                   // [3][H][D]
  __bf16* qbf = (__bf16*)(ws + (size_t)3 * HH * DD * 2);       // [B*T][H]
  __bf16* kbf = qbf + (size_t)BB * TT * HH;                    // [B*T][H]
  __bf16* vtb = kbf + (size_t)BB * TT * HH;                    // [B][H][T]

  prep_weights<<<(3 * HH * DD) / 256, 256, 0, stream>>>(Wk, Wq, Wv, wt);
  qkv_proj<<<(BB * TT / 16) / 8, 256, 0, stream>>>(x, wt, qbf, kbf, vtb);
  flash_attn<<<BB * TT / 16, 32 * NW, 0, stream>>>(qbf, kbf, vtb, out);
}
